// MaskedCrossAttention_47150150975842
// MI455X (gfx1250) — compile-verified, matrix operand order checked by reading, not run on hardware
//
#include <hip/hip_runtime.h>
#include <stdint.h>

typedef __attribute__((ext_vector_type(16))) _Float16 v16h;
typedef __attribute__((ext_vector_type(8)))  float    v8f;

namespace {
constexpr int kB    = 4;
constexpr int kN    = 2048;          // text tokens per batch
constexpr int kT    = 8;
constexpr int kNL   = 64;
constexpr int kTN   = kT * kNL;      // 512 media tokens per batch
constexpr int kD    = 1024;
constexpr int kH    = 16;
constexpr int kHD   = 64;
constexpr int kRows = kB * kN;       // 8192 text rows
constexpr int kMRows = kB * kTN;     // 2048 media rows

constexpr int KB  = 32;              // K chunk per LDS stage
constexpr int LDH = 40;              // LDS row pitch in halves (pad vs 32)
}

union F16frag { v16h v; uint32_t u[8]; };
union Half8   { uint4 u; _Float16 h[8]; };

// ---------------------------------------------------------------------------
// f32 -> f16 with transpose: src is [K][N] row-major, dst is [N][K] row-major.
__global__ void convt_kernel(const float* __restrict__ src, _Float16* __restrict__ dst,
                             int K, int N) {
    int idx = blockIdx.x * 256 + threadIdx.x;     // grid sized exactly K*N/256
    int k = idx / N;
    int n = idx - k * N;
    dst[(size_t)n * K + k] = (_Float16)src[idx];
}

// f32 -> f16 straight copy
__global__ void conv_kernel(const float* __restrict__ src, _Float16* __restrict__ dst, int n) {
    int idx = blockIdx.x * 256 + threadIdx.x;
    if (idx < n) dst[idx] = (_Float16)src[idx];
}

// ---------------------------------------------------------------------------
// Inclusive cumsum of media_locations (bool, 1 byte) per batch row -> int times
__global__ void cumsum_kernel(const unsigned char* __restrict__ loc, int* __restrict__ times) {
    int b = blockIdx.x;
    int t = threadIdx.x;
    __shared__ int part[256];
    const unsigned char* row = loc + (size_t)b * kN;
    int base = t * 8;
    int v[8];
    int s = 0;
#pragma unroll
    for (int i = 0; i < 8; i++) { v[i] = row[base + i] ? 1 : 0; s += v[i]; }
    part[t] = s;
    __syncthreads();
    if (t == 0) {
        int acc = 0;
        for (int i = 0; i < 256; i++) { int x = part[i]; part[i] = acc; acc += x; }  // exclusive
    }
    __syncthreads();
    int acc = part[t];
    int* orow = times + (size_t)b * kN + base;
#pragma unroll
    for (int i = 0; i < 8; i++) { acc += v[i]; orow[i] = acc; }
}

// ---------------------------------------------------------------------------
// LayerNorm over last dim (1024), write f16. One 256-thread block per row.
__global__ void ln_kernel(const float* __restrict__ text, const float* __restrict__ g,
                          const float* __restrict__ bb, _Float16* __restrict__ tnh) {
    int row = blockIdx.x;
    int t = threadIdx.x;
    const float* x = text + (size_t)row * kD;
    float4 xv = reinterpret_cast<const float4*>(x)[t];
    float s  = xv.x + xv.y + xv.z + xv.w;
    float s2 = xv.x * xv.x + xv.y * xv.y + xv.z * xv.z + xv.w * xv.w;
    __shared__ float rs[256], rq[256];
    rs[t] = s; rq[t] = s2;
    __syncthreads();
    for (int st = 128; st > 0; st >>= 1) {
        if (t < st) { rs[t] += rs[t + st]; rq[t] += rq[t + st]; }
        __syncthreads();
    }
    float mean = rs[0] * (1.0f / kD);
    float var  = rq[0] * (1.0f / kD) - mean * mean;
    float rstd = rsqrtf(var + 1e-5f);
    float4 gv = reinterpret_cast<const float4*>(g)[t];
    float4 bv = reinterpret_cast<const float4*>(bb)[t];
    union { uint2 u; _Float16 h[4]; } o;
    o.h[0] = (_Float16)((xv.x - mean) * rstd * gv.x + bv.x);
    o.h[1] = (_Float16)((xv.y - mean) * rstd * gv.y + bv.y);
    o.h[2] = (_Float16)((xv.z - mean) * rstd * gv.z + bv.z);
    o.h[3] = (_Float16)((xv.w - mean) * rstd * gv.w + bv.w);
    reinterpret_cast<uint2*>(tnh + (size_t)row * kD)[t] = o.u;
}

// ---------------------------------------------------------------------------
// Tiled f16 WMMA GEMM: C[M,Nn] = A[M,K] * Bt[Nn,K]^T (Bt stored transposed).
// Block = 128 threads (4 waves), 64x64 block tile, 32x32 per wave.
// Staging via GLOBAL_LOAD_ASYNC_TO_LDS_B128 (ASYNCcnt) with LDS double buffer.
// If Cf != nullptr: f32 output (+ optional bias). Else f16 output to Ch.
__global__ __launch_bounds__(128)
void gemm_kernel(const _Float16* __restrict__ A, const _Float16* __restrict__ Bt,
                 int M, int Nn, int K,
                 _Float16* __restrict__ Ch, float* __restrict__ Cf,
                 const float* __restrict__ bias) {
    __shared__ _Float16 Ash[2][64 * LDH];
    __shared__ _Float16 Bsh[2][64 * LDH];

    const int tid  = threadIdx.x;
    const int lane = tid & 31;
    const int wave = tid >> 5;
    const int wm = wave >> 1;          // 0..1
    const int wn = wave & 1;           // 0..1
    const int nbase = blockIdx.x * 64;
    const int mbase = blockIdx.y * 64;

    // Cooperative staging: thread t owns one 32B (two b128) segment of one A
    // row and one B row per K chunk.
    const int r   = tid >> 1;          // 0..63 (row within tile)
    const int seg = tid & 1;           // 0..1  (16-half segment)
    const _Float16* agp = A  + (size_t)(mbase + r) * K + seg * 16;
    const _Float16* bgp = Bt + (size_t)(nbase + r) * K + seg * 16;
    const uint32_t soffB = (uint32_t)(r * LDH + seg * 16) * sizeof(_Float16);
    // Workgroup-relative LDS byte offsets (low 32 bits of the generic pointer
    // carry the LDS offset; aperture bits live in [63:32]).
    uint32_t ldsA[2], ldsB[2];
    ldsA[0] = (uint32_t)(uintptr_t)(&Ash[0][0]) + soffB;
    ldsA[1] = (uint32_t)(uintptr_t)(&Ash[1][0]) + soffB;
    ldsB[0] = (uint32_t)(uintptr_t)(&Bsh[0][0]) + soffB;
    ldsB[1] = (uint32_t)(uintptr_t)(&Bsh[1][0]) + soffB;

    // Per-lane fragment addressing (ISA 7.12.2, 16-bit operands):
    // lanes 0-15 hold K pairs {0..7,16..23}; lanes 16-31 hold {8..15,24..31}.
    const int lrow = lane & 15;
    const int kp   = (lane >> 4) << 2;          // u32 pair base: 0 or 4

    v8f acc[2][2];
#pragma unroll
    for (int a = 0; a < 2; a++)
#pragma unroll
        for (int b2 = 0; b2 < 2; b2++)
#pragma unroll
            for (int e = 0; e < 8; e++) acc[a][b2][e] = 0.0f;

    // Issue one K-chunk of async global->LDS copies (4 x b128 per thread).
    // INST_OFFSET applies to both LDS and global addresses (ISA 08 §4.4).
    auto issue_chunk = [&](int buf, int kb) {
        asm volatile(
            "global_load_async_to_lds_b128 %0, %2, off\n\t"
            "global_load_async_to_lds_b128 %0, %2, off offset:16\n\t"
            "global_load_async_to_lds_b128 %1, %3, off\n\t"
            "global_load_async_to_lds_b128 %1, %3, off offset:16"
            :: "v"(ldsA[buf]), "v"(ldsB[buf]), "v"(agp + kb), "v"(bgp + kb)
            : "memory");
    };

    const int nchunks = K / KB;
    issue_chunk(0, 0);

    for (int c = 0; c < nchunks; c++) {
        const int buf = c & 1;
        // Our async writes for chunk c are done; barrier makes every wave's
        // writes visible (and orders prior reads of buf^1 before the writes
        // we are about to issue into it).
        asm volatile("s_wait_asynccnt 0x0" ::: "memory");
        __syncthreads();
        if (c + 1 < nchunks) issue_chunk(buf ^ 1, (c + 1) * KB);

        F16frag af[2], bf[2];
#pragma unroll
        for (int mt = 0; mt < 2; mt++) {
            const uint32_t* p = reinterpret_cast<const uint32_t*>(Ash[buf])
                              + (wm * 32 + mt * 16 + lrow) * (LDH / 2) + kp;
#pragma unroll
            for (int rr = 0; rr < 4; rr++) { af[mt].u[rr] = p[rr]; af[mt].u[4 + rr] = p[8 + rr]; }
        }
#pragma unroll
        for (int nt = 0; nt < 2; nt++) {
            const uint32_t* p = reinterpret_cast<const uint32_t*>(Bsh[buf])
                              + (wn * 32 + nt * 16 + lrow) * (LDH / 2) + kp;
#pragma unroll
            for (int rr = 0; rr < 4; rr++) { bf[nt].u[rr] = p[rr]; bf[nt].u[4 + rr] = p[8 + rr]; }
        }

#pragma unroll
        for (int mt = 0; mt < 2; mt++)
#pragma unroll
            for (int nt = 0; nt < 2; nt++)
                acc[mt][nt] = __builtin_amdgcn_wmma_f32_16x16x32_f16(
                    false, af[mt].v, false, bf[nt].v, (short)0, acc[mt][nt], false, false);
    }

    // Epilogue: C/D layout — lane holds n = lane&15, rows (lane>=16 ? 8:0)+rr.
    const int mrow0 = (lane >> 4) << 3;
#pragma unroll
    for (int mt = 0; mt < 2; mt++) {
#pragma unroll
        for (int nt = 0; nt < 2; nt++) {
            const int ng = nbase + wn * 32 + nt * 16 + lrow;
#pragma unroll
            for (int rr = 0; rr < 8; rr++) {
                const int mg = mbase + wm * 32 + mt * 16 + mrow0 + rr;
                if (Cf) {
                    float bsv = bias ? bias[ng] : 0.0f;
                    Cf[(size_t)mg * Nn + ng] = acc[mt][nt][rr] + bsv;
                } else {
                    Ch[(size_t)mg * Nn + ng] = (_Float16)acc[mt][nt][rr];
                }
            }
        }
    }
}

// ---------------------------------------------------------------------------
// Masked cross-attention: each query attends to exactly one 64-key media block
// (block = text_time-1), or outputs zero if text_time==0.
// One wave32 per (b, h, i). Lane L owns keys 2L, 2L+1 for scores; dims L, L+32
// for the output accumulation.
__global__ __launch_bounds__(256)
void attn_kernel(const _Float16* __restrict__ qh, const _Float16* __restrict__ kvh,
                 const int* __restrict__ ttimes, _Float16* __restrict__ oh) {
    const int lane = threadIdx.x & 31;
    const int wave = threadIdx.x >> 5;
    const int gw = blockIdx.x * 8 + wave;          // (b*H + h)*N + i
    const int i = gw & (kN - 1);
    const int h = (gw >> 11) & (kH - 1);
    const int b = gw >> 15;

    _Float16* orow = oh + (size_t)(b * kN + i) * kD + h * kHD;
    const int tt = ttimes[(size_t)b * kN + i];
    if (tt == 0) {
        reinterpret_cast<uint32_t*>(orow)[lane] = 0u;   // 64 zero halves
        return;
    }
    const int j0 = (tt - 1) * kNL;

    const _Float16* qrow = qh + (size_t)(b * kN + i) * kD + h * kHD;
    const _Float16* k0 = kvh + (size_t)(b * kTN + j0 + 2 * lane) * (2 * kD) + h * kHD;
    const _Float16* k1 = k0 + 2 * kD;

    float s0 = 0.0f, s1 = 0.0f;
#pragma unroll
    for (int d0 = 0; d0 < kHD; d0 += 8) {
        Half8 qv, kv0, kv1;
        qv.u  = *reinterpret_cast<const uint4*>(qrow + d0);
        kv0.u = *reinterpret_cast<const uint4*>(k0 + d0);
        kv1.u = *reinterpret_cast<const uint4*>(k1 + d0);
#pragma unroll
        for (int e = 0; e < 8; e++) {
            float q = (float)qv.h[e];
            s0 += q * (float)kv0.h[e];
            s1 += q * (float)kv1.h[e];
        }
    }

    // wave-wide softmax over 64 scores (2 per lane), wave32 shuffles
    float m = fmaxf(s0, s1);
#pragma unroll
    for (int off = 16; off > 0; off >>= 1) m = fmaxf(m, __shfl_xor(m, off, 32));
    float e0 = __expf(s0 - m), e1 = __expf(s1 - m);
    float sum = e0 + e1;
#pragma unroll
    for (int off = 16; off > 0; off >>= 1) sum += __shfl_xor(sum, off, 32);
    float inv = 1.0f / sum;
    float a0 = e0 * inv, a1 = e1 * inv;

    // out[d] = sum_j a_j * v[j][d]; lane owns dims lane and lane+32
    const _Float16* vbase = kvh + (size_t)(b * kTN + j0) * (2 * kD) + kD + h * kHD;
    float acc0 = 0.0f, acc1 = 0.0f;
#pragma unroll
    for (int jj = 0; jj < 32; jj++) {
        float aA = __shfl(a0, jj, 32);      // weight of key 2*jj
        float aB = __shfl(a1, jj, 32);      // weight of key 2*jj+1
        const _Float16* v0 = vbase + (size_t)(2 * jj) * (2 * kD);
        const _Float16* v1 = v0 + 2 * kD;
        acc0 += aA * (float)v0[lane]      + aB * (float)v1[lane];
        acc1 += aA * (float)v0[lane + 32] + aB * (float)v1[lane + 32];
    }
    orow[lane]      = (_Float16)acc0;
    orow[lane + 32] = (_Float16)acc1;
}

// ---------------------------------------------------------------------------
extern "C" void kernel_launch(void* const* d_in, const int* in_sizes, int n_in,
                              void* d_out, int out_size, void* d_ws, size_t ws_size,
                              hipStream_t stream) {
    (void)in_sizes; (void)n_in; (void)out_size; (void)ws_size;
    const float* text  = (const float*)d_in[0];
    const float* media = (const float*)d_in[1];
    const unsigned char* mloc = (const unsigned char*)d_in[2];  // bool array
    const float* wq  = (const float*)d_in[3];
    const float* wkv = (const float*)d_in[4];
    const float* wo  = (const float*)d_in[5];
    const float* bo  = (const float*)d_in[6];
    const float* lng = (const float*)d_in[7];
    const float* lnb = (const float*)d_in[8];
    float* out = (float*)d_out;

    char* p = (char*)d_ws;
    auto alloc = [&](size_t bytes) -> char* {
        char* r = p; p += (bytes + 255) & ~(size_t)255; return r;
    };
    _Float16* wq_t  = (_Float16*)alloc((size_t)kD * kD * 2);          // [1024][1024]
    _Float16* wkv_t = (_Float16*)alloc((size_t)(2 * kD) * kD * 2);    // [2048][1024]
    _Float16* wo_t  = (_Float16*)alloc((size_t)kD * kD * 2);          // [1024][1024]
    _Float16* med_h = (_Float16*)alloc((size_t)kMRows * kD * 2);      // [2048][1024]
    _Float16* tn_h  = (_Float16*)alloc((size_t)kRows * kD * 2);       // [8192][1024]
    _Float16* q_h   = (_Float16*)alloc((size_t)kRows * kD * 2);       // [8192][1024]
    _Float16* kv_h  = (_Float16*)alloc((size_t)kMRows * 2 * kD * 2);  // [2048][2048]
    _Float16* at_h  = (_Float16*)alloc((size_t)kRows * kD * 2);       // [8192][1024]
    int*      times = (int*)alloc((size_t)kRows * 4);

    // 1. weights -> f16 transposed [N][K]; media -> f16
    convt_kernel<<<kD * kD / 256, 256, 0, stream>>>(wq, wq_t, kD, kD);
    convt_kernel<<<kD * 2 * kD / 256, 256, 0, stream>>>(wkv, wkv_t, kD, 2 * kD);
    convt_kernel<<<kD * kD / 256, 256, 0, stream>>>(wo, wo_t, kD, kD);
    conv_kernel<<<kMRows * kD / 256, 256, 0, stream>>>(media, med_h, kMRows * kD);

    // 2. text_times = cumsum(media_locations)
    cumsum_kernel<<<kB, 256, 0, stream>>>(mloc, times);

    // 3. LayerNorm -> f16
    ln_kernel<<<kRows, 256, 0, stream>>>(text, lng, lnb, tn_h);

    // 4. q = tn @ wq  (8192x1024x1024)
    gemm_kernel<<<dim3(kD / 64, kRows / 64), 128, 0, stream>>>(
        tn_h, wq_t, kRows, kD, kD, q_h, nullptr, nullptr);

    // 5. kv = media @ wkv  (2048x2048x1024); cols [0:1024)=K, [1024:2048)=V
    gemm_kernel<<<dim3(2 * kD / 64, kMRows / 64), 128, 0, stream>>>(
        med_h, wkv_t, kMRows, 2 * kD, kD, kv_h, nullptr, nullptr);

    // 6. masked single-block attention
    attn_kernel<<<(kB * kH * kN) / 8, 256, 0, stream>>>(q_h, kv_h, times, at_h);

    // 7. out = attn_out @ wo + bo  (f32 output)
    gemm_kernel<<<dim3(kD / 64, kRows / 64), 128, 0, stream>>>(
        at_h, wo_t, kRows, kD, kD, nullptr, out, bo);
}
